// TrashPredictor_89953795047696
// MI455X (gfx1250) — matrix operand order repaired and checked by reading, not compiled, not run on hardware
//
#include <hip/hip_runtime.h>

#define N_NODES 100000
#define N_EDGES 1600000
#define F_NODE  16
#define H_LSTM  64
#define T_STEPS 168
#define C_GAT   128
#define HEADS   4
#define FC_HID  64

typedef __attribute__((ext_vector_type(16))) _Float16 v16h;
typedef __attribute__((ext_vector_type(8)))  float    v8f;

__device__ __forceinline__ float sigmoidf_(float x) { return 1.f / (1.f + __expf(-x)); }

// monotonic float<->uint mapping for lock-free atomic max on f32
__device__ __forceinline__ unsigned fenc(float f) {
    unsigned u = __float_as_uint(f);
    return (u & 0x80000000u) ? ~u : (u | 0x80000000u);
}
__device__ __forceinline__ float fdec(unsigned u) {
    u = (u & 0x80000000u) ? (u & 0x7FFFFFFFu) : ~u;
    return __uint_as_float(u);
}

// ---------------------------------------------------------------------------
// 2-layer LSTM over T steps, single block of 256 threads (8 wave32).
// Also folds the constant ts-embedding contribution of GAT layer 1 into a
// 128-wide bias vector: bias1[c] = sum_k h1[k] * W1[16+k][c].
// ---------------------------------------------------------------------------
__global__ void lstm_kernel(const float* __restrict__ ts,
                            const float* __restrict__ Wih0, const float* __restrict__ Whh0,
                            const float* __restrict__ b0,
                            const float* __restrict__ Wih1, const float* __restrict__ Whh1,
                            const float* __restrict__ b1,
                            const float* __restrict__ W1,
                            float* __restrict__ ts_emb, float* __restrict__ bias1) {
    __shared__ float h0[H_LSTM], c0[H_LSTM], h1[H_LSTM], c1[H_LSTM], g[4 * H_LSTM];
    const int t = threadIdx.x;  // 0..255, one gate element each
    if (t < H_LSTM) { h0[t] = 0.f; c0[t] = 0.f; h1[t] = 0.f; c1[t] = 0.f; }
    __syncthreads();
    for (int step = 0; step < T_STEPS; ++step) {
        const float x0 = ts[step * 3 + 0], x1 = ts[step * 3 + 1], x2 = ts[step * 3 + 2];
        float acc = Wih0[t * 3 + 0] * x0 + Wih0[t * 3 + 1] * x1 + Wih0[t * 3 + 2] * x2 + b0[t];
        for (int j = 0; j < H_LSTM; ++j) acc += Whh0[t * H_LSTM + j] * h0[j];
        g[t] = acc;
        __syncthreads();
        if (t < H_LSTM) {
            float i = sigmoidf_(g[t]), f = sigmoidf_(g[H_LSTM + t]);
            float gg = tanhf(g[2 * H_LSTM + t]), o = sigmoidf_(g[3 * H_LSTM + t]);
            float c = f * c0[t] + i * gg;
            c0[t] = c; h0[t] = o * tanhf(c);
        }
        __syncthreads();
        acc = b1[t];
        for (int j = 0; j < H_LSTM; ++j) acc += Wih1[t * H_LSTM + j] * h0[j];
        for (int j = 0; j < H_LSTM; ++j) acc += Whh1[t * H_LSTM + j] * h1[j];
        g[t] = acc;
        __syncthreads();
        if (t < H_LSTM) {
            float i = sigmoidf_(g[t]), f = sigmoidf_(g[H_LSTM + t]);
            float gg = tanhf(g[2 * H_LSTM + t]), o = sigmoidf_(g[3 * H_LSTM + t]);
            float c = f * c1[t] + i * gg;
            c1[t] = c; h1[t] = o * tanhf(c);
        }
        __syncthreads();
    }
    if (t < H_LSTM) ts_emb[t] = h1[t];
    __syncthreads();
    if (t < C_GAT) {
        float s = 0.f;
        for (int k = 0; k < H_LSTM; ++k) s += h1[k] * W1[(F_NODE + k) * C_GAT + t];
        bias1[t] = s;
    }
}

// ---------------------------------------------------------------------------
// Pack B (K x Nout, f32 row-major) into f16 WMMA B-fragment layout:
//   Bp[((kt*ntiles + nt)*32 + lane)*16 + j]
// lane%16 = column-in-tile, lane/16 selects K 0..15 vs 16..31 of the k-tile,
// j = contiguous K within the half. Zero-pads kk >= K.
// ---------------------------------------------------------------------------
__global__ void pack_b(const float* __restrict__ B, int K, int Nout, int ktiles,
                       _Float16* __restrict__ Bp) {
    const int ntiles = Nout >> 4;
    const int total = ktiles * ntiles * 512;
    for (int idx = blockIdx.x * blockDim.x + threadIdx.x; idx < total;
         idx += gridDim.x * blockDim.x) {
        const int j = idx & 15;
        const int lane = (idx >> 4) & 31;
        const int nt = (idx >> 9) % ntiles;
        const int kt = idx / (ntiles << 9);
        const int col = nt * 16 + (lane & 15);
        const int kk = kt * 32 + ((lane >> 4) << 4) + j;
        Bp[idx] = (kk < K) ? (_Float16)B[(size_t)kk * Nout + col] : (_Float16)0.f;
    }
}

// ---------------------------------------------------------------------------
// WMMA GEMM: Cout(M x NOUT) = A(M x KREAL, f32) * Bpacked [+bias][relu]
// Block = 256 thr (8 wave32): 16 rows x NOUT cols; wave w owns cols [16w,16w+16).
// A tile staged once in LDS in fragment order (f16); K fully unrolled.
// All shape/epilogue parameters are compile time: branch-free epilogue,
// immediate-offset stores. M must be a multiple of 16.
// ---------------------------------------------------------------------------
template <int KTILES, int KREAL, int NOUT, int BIAS, int RELU>
__global__ __launch_bounds__(256) void gemm_wmma(const float* __restrict__ A, int lda,
                                                 const _Float16* __restrict__ Bp,
                                                 const float* __restrict__ bias,
                                                 float* __restrict__ Cout) {
    __shared__ _Float16 lds_a[KTILES * 512];  // KTILES x (32 lanes x 16 halfs)
    const int m0 = blockIdx.x * 16;
    // cooperative, coalesced fill of the 16 x (32*KTILES) A tile, pre-swizzled
    #pragma unroll
    for (int e = threadIdx.x; e < KTILES * 512; e += 256) {
        const int kt = e >> 9, r = (e >> 5) & 15, cc = e & 31;
        const int kk = kt * 32 + cc;
        float v = 0.f;
        if ((KREAL & 31) == 0 || kk < KREAL) v = A[(size_t)(m0 + r) * lda + kk];
        const int lane = r + ((cc >> 4) << 4);
        const int j = cc & 15;
        lds_a[(kt * 32 + lane) * 16 + j] = (_Float16)v;
    }
    __syncthreads();
    const int wave = threadIdx.x >> 5;
    const int lane = threadIdx.x & 31;
    const int n0 = wave * 16;
    if (n0 >= NOUT) return;  // after the barrier; no further block sync below
    constexpr int NTILES = NOUT >> 4;
    v8f c = {};
    #pragma unroll
    for (int kt = 0; kt < KTILES; ++kt) {
        v16h a = *(const v16h*)(lds_a + (kt * 32 + lane) * 16);
        v16h b = *(const v16h*)(Bp + ((size_t)(kt * NTILES + wave) * 32 + lane) * 16);
        c = __builtin_amdgcn_wmma_f32_16x16x32_f16(false, a, false, b, (short)0, c, false, false);
    }
    const int rlo = lane & 15, hi = lane >> 4;
    const int col = n0 + rlo;
    float bv = 0.f;
    if (BIAS) bv = bias[col];                     // one coalesced load, loop-invariant
    float* cp = Cout + (size_t)(m0 + hi * 8) * NOUT + col;
    #pragma unroll
    for (int r = 0; r < 8; ++r) {
        float v = c[r];
        if (BIAS) v += bv;
        if (RELU) v = fmaxf(v, 0.f);
        cp[(size_t)r * NOUT] = v;                 // immediate-offset b32 stores
    }
}

// a_src/a_dst = Wx @ A[:C] / Wx @ A[C:]  (N x 4 each)
__global__ void attn_proj(const float* __restrict__ Wx, const float* __restrict__ Aatt,
                          float4* __restrict__ a_src, float4* __restrict__ a_dst, int n) {
    const int i = blockIdx.x * blockDim.x + threadIdx.x;
    if (i >= n) return;
    float s[HEADS] = {0.f, 0.f, 0.f, 0.f}, d[HEADS] = {0.f, 0.f, 0.f, 0.f};
    const float* w = Wx + (size_t)i * C_GAT;
    for (int c = 0; c < C_GAT; ++c) {
        const float v = w[c];
        const float4 as = ((const float4*)Aatt)[c];
        const float4 ad = ((const float4*)Aatt)[C_GAT + c];
        s[0] += v * as.x; s[1] += v * as.y; s[2] += v * as.z; s[3] += v * as.w;
        d[0] += v * ad.x; d[1] += v * ad.y; d[2] += v * ad.z; d[3] += v * ad.w;
    }
    a_src[i] = make_float4(s[0], s[1], s[2], s[3]);
    a_dst[i] = make_float4(d[0], d[1], d[2], d[3]);
}

__device__ __forceinline__ void edge_e4(const float4& as, const float4& ad, float* eh) {
    eh[0] = as.x + ad.x; eh[1] = as.y + ad.y; eh[2] = as.z + ad.z; eh[3] = as.w + ad.w;
    #pragma unroll
    for (int h = 0; h < HEADS; ++h) eh[h] = (eh[h] > 0.f) ? eh[h] : 0.2f * eh[h];
}

// global per-head max of leaky_relu(a_src[src]+a_dst[dst]) over all edges
__global__ void edge_max(const int* __restrict__ src, const int* __restrict__ dst,
                         const float4* __restrict__ a_src, const float4* __restrict__ a_dst,
                         unsigned* __restrict__ hmax) {
    float m[HEADS] = {-3.402823466e38f, -3.402823466e38f, -3.402823466e38f, -3.402823466e38f};
    const int stride = gridDim.x * blockDim.x;
    for (int e = blockIdx.x * blockDim.x + threadIdx.x; e < N_EDGES; e += stride) {
        float eh[HEADS];
        edge_e4(a_src[src[e]], a_dst[dst[e]], eh);
        #pragma unroll
        for (int h = 0; h < HEADS; ++h) m[h] = fmaxf(m[h], eh[h]);
    }
    __shared__ float red[256][HEADS];
    #pragma unroll
    for (int h = 0; h < HEADS; ++h) red[threadIdx.x][h] = m[h];
    __syncthreads();
    for (int s = 128; s > 0; s >>= 1) {
        if (threadIdx.x < s) {
            #pragma unroll
            for (int h = 0; h < HEADS; ++h)
                red[threadIdx.x][h] = fmaxf(red[threadIdx.x][h], red[threadIdx.x + s][h]);
        }
        __syncthreads();
    }
    if (threadIdx.x < HEADS) atomicMax(&hmax[threadIdx.x], fenc(red[0][threadIdx.x]));
}

// denom[dst][h] += exp(e - max_h)
__global__ void edge_expsum(const int* __restrict__ src, const int* __restrict__ dst,
                            const float4* __restrict__ a_src, const float4* __restrict__ a_dst,
                            const unsigned* __restrict__ hmax, float* __restrict__ denom) {
    float mh[HEADS];
    #pragma unroll
    for (int h = 0; h < HEADS; ++h) mh[h] = fdec(hmax[h]);
    const int stride = gridDim.x * blockDim.x;
    for (int e = blockIdx.x * blockDim.x + threadIdx.x; e < N_EDGES; e += stride) {
        const int d = dst[e];
        float eh[HEADS];
        edge_e4(a_src[src[e]], a_dst[d], eh);
        #pragma unroll
        for (int h = 0; h < HEADS; ++h)
            atomicAdd(&denom[d * HEADS + h], __expf(eh[h] - mh[h]));
    }
}

// one wave32 per edge: acc[dst] += Wx[src] * coef ; 4 f32 atomics per lane.
// ref: denom = segsum + 1e-9, then +1e-9 again at use -> +2e-9 total.
__global__ void edge_accum(const int* __restrict__ src, const int* __restrict__ dst,
                           const float4* __restrict__ a_src, const float4* __restrict__ a_dst,
                           const unsigned* __restrict__ hmax, const float4* __restrict__ denom,
                           const float* __restrict__ Wx, float* __restrict__ acc) {
    const int e = (int)((blockIdx.x * (unsigned)blockDim.x + threadIdx.x) >> 5);
    const int lane = threadIdx.x & 31;
    if (e >= N_EDGES) return;
    const int s = src[e], d = dst[e];
    const float* w = Wx + (size_t)s * C_GAT;
    __builtin_prefetch(w + lane, 0, 1);  // global_prefetch_b8 on the gathered row
    float eh[HEADS];
    edge_e4(a_src[s], a_dst[d], eh);
    const float4 dn = denom[d];
    float coef = __expf(eh[0] - fdec(hmax[0])) / (dn.x + 2e-9f)
               + __expf(eh[1] - fdec(hmax[1])) / (dn.y + 2e-9f)
               + __expf(eh[2] - fdec(hmax[2])) / (dn.z + 2e-9f)
               + __expf(eh[3] - fdec(hmax[3])) / (dn.w + 2e-9f);
    float* o = acc + (size_t)d * C_GAT;
    #pragma unroll
    for (int j = 0; j < 4; ++j) {
        const int col = lane + 32 * j;
        atomicAdd(&o[col], w[col] * coef);
    }
}

// x = elu(acc / HEADS), in place over N*C elements
__global__ void node_elu(float* __restrict__ p, long n) {
    const long stride = (long)gridDim.x * blockDim.x;
    for (long i = (long)blockIdx.x * blockDim.x + threadIdx.x; i < n; i += stride) {
        const float v = p[i] * 0.25f;
        p[i] = (v > 0.f) ? v : (__expf(v) - 1.f);
    }
}

__global__ void fc2_kernel(const float* __restrict__ h, const float* __restrict__ w,
                           const float* __restrict__ b, float* __restrict__ out, int n) {
    const int i = blockIdx.x * blockDim.x + threadIdx.x;
    if (i >= n) return;
    const float4* hp = (const float4*)(h + (size_t)i * FC_HID);
    const float4* wp = (const float4*)w;
    float s = b[0];
    #pragma unroll
    for (int j = 0; j < FC_HID / 4; ++j) {
        const float4 hv = hp[j], wv = wp[j];
        s += hv.x * wv.x + hv.y * wv.y + hv.z * wv.z + hv.w * wv.w;
    }
    out[i] = s;
}

__global__ void fill_f32(float* __restrict__ p, long n, float v) {
    const long stride = (long)gridDim.x * blockDim.x;
    for (long i = (long)blockIdx.x * blockDim.x + threadIdx.x; i < n; i += stride) p[i] = v;
}
__global__ void fill_u32(unsigned* __restrict__ p, int n, unsigned v) {
    const int i = blockIdx.x * blockDim.x + threadIdx.x;
    if (i < n) p[i] = v;
}

extern "C" void kernel_launch(void* const* d_in, const int* in_sizes, int n_in,
                              void* d_out, int out_size, void* d_ws, size_t ws_size,
                              hipStream_t stream) {
    (void)in_sizes; (void)n_in; (void)out_size; (void)ws_size;
    const float* node_feat = (const float*)d_in[0];
    const int*   edge_idx  = (const int*)d_in[1];
    const float* ts_seq    = (const float*)d_in[2];
    const float* Wih0 = (const float*)d_in[3];
    const float* Whh0 = (const float*)d_in[4];
    const float* b0   = (const float*)d_in[5];
    const float* Wih1 = (const float*)d_in[6];
    const float* Whh1 = (const float*)d_in[7];
    const float* b1   = (const float*)d_in[8];
    const float* W1   = (const float*)d_in[9];
    const float* A1   = (const float*)d_in[10];
    const float* W2   = (const float*)d_in[11];
    const float* A2   = (const float*)d_in[12];
    const float* fc1w = (const float*)d_in[13];
    const float* fc1b = (const float*)d_in[14];
    const float* fc2w = (const float*)d_in[15];
    const float* fc2b = (const float*)d_in[16];
    float* out = (float*)d_out;

    // workspace layout (floats)
    float*    ws     = (float*)d_ws;
    float*    ts_emb = ws;                                     // 64
    float*    bias1  = ws + 64;                                // 128
    unsigned* hmax   = (unsigned*)(ws + 192);                  // 4
    float*    a_src  = ws + 256;                               // N*4 (16B aligned)
    float*    a_dst  = a_src + (size_t)N_NODES * HEADS;        // N*4
    float*    denom  = a_dst + (size_t)N_NODES * HEADS;        // N*4
    float*    bufA   = denom + (size_t)N_NODES * HEADS;        // N*128 (Wx)
    float*    bufB   = bufA + (size_t)N_NODES * C_GAT;         // N*128 (acc / x)
    _Float16* W1p    = (_Float16*)(bufB + (size_t)N_NODES * C_GAT);  // 1*8*512
    _Float16* W2p    = W1p + 4096;                             // 4*8*512
    _Float16* fc1p   = W2p + 16384;                            // 4*4*512

    const int* src = edge_idx;
    const int* dst = edge_idx + N_EDGES;
    const int mblocks = N_NODES / 16;                          // 6250, exact
    const int nblocks = (N_NODES + 255) / 256;

    // ---- weight packing (fragment layout, f16) + LSTM ----
    pack_b<<<16, 256, 0, stream>>>(W1, F_NODE, C_GAT, 1, W1p);   // rows 0..15 of W1
    pack_b<<<64, 256, 0, stream>>>(W2, C_GAT, C_GAT, 4, W2p);
    pack_b<<<32, 256, 0, stream>>>(fc1w, C_GAT, FC_HID, 4, fc1p);
    lstm_kernel<<<1, 256, 0, stream>>>(ts_seq, Wih0, Whh0, b0, Wih1, Whh1, b1, W1, ts_emb, bias1);

    // ---- GAT layer 1 ----
    // Wx1 = node_feat @ W1[:16] + bias1   (ts columns are constant across nodes)
    gemm_wmma<1, F_NODE, C_GAT, 1, 0><<<mblocks, 256, 0, stream>>>(node_feat, F_NODE, W1p, bias1, bufA);
    attn_proj<<<nblocks, 256, 0, stream>>>(bufA, A1, (float4*)a_src, (float4*)a_dst, N_NODES);
    fill_u32<<<1, 32, 0, stream>>>(hmax, HEADS, 0u);
    fill_f32<<<2048, 256, 0, stream>>>(denom, (long)N_NODES * HEADS, 0.f);
    fill_f32<<<8192, 256, 0, stream>>>(bufB, (long)N_NODES * C_GAT, 0.f);
    edge_max<<<1024, 256, 0, stream>>>(src, dst, (const float4*)a_src, (const float4*)a_dst, hmax);
    edge_expsum<<<4096, 256, 0, stream>>>(src, dst, (const float4*)a_src, (const float4*)a_dst, hmax, denom);
    edge_accum<<<(N_EDGES + 7) / 8, 256, 0, stream>>>(src, dst, (const float4*)a_src, (const float4*)a_dst,
                                                      hmax, (const float4*)denom, bufA, bufB);
    node_elu<<<8192, 256, 0, stream>>>(bufB, (long)N_NODES * C_GAT);

    // ---- GAT layer 2 ----
    gemm_wmma<4, C_GAT, C_GAT, 0, 0><<<mblocks, 256, 0, stream>>>(bufB, C_GAT, W2p, nullptr, bufA);
    attn_proj<<<nblocks, 256, 0, stream>>>(bufA, A2, (float4*)a_src, (float4*)a_dst, N_NODES);
    fill_u32<<<1, 32, 0, stream>>>(hmax, HEADS, 0u);
    fill_f32<<<2048, 256, 0, stream>>>(denom, (long)N_NODES * HEADS, 0.f);
    fill_f32<<<8192, 256, 0, stream>>>(bufB, (long)N_NODES * C_GAT, 0.f);   // x2 dead after GEMM2
    edge_max<<<1024, 256, 0, stream>>>(src, dst, (const float4*)a_src, (const float4*)a_dst, hmax);
    edge_expsum<<<4096, 256, 0, stream>>>(src, dst, (const float4*)a_src, (const float4*)a_dst, hmax, denom);
    edge_accum<<<(N_EDGES + 7) / 8, 256, 0, stream>>>(src, dst, (const float4*)a_src, (const float4*)a_dst,
                                                      hmax, (const float4*)denom, bufA, bufB);
    node_elu<<<8192, 256, 0, stream>>>(bufB, (long)N_NODES * C_GAT);

    // ---- FC head ----
    gemm_wmma<4, C_GAT, FC_HID, 1, 1><<<mblocks, 256, 0, stream>>>(bufB, C_GAT, fc1p, fc1b, bufA);
    fc2_kernel<<<nblocks, 256, 0, stream>>>(bufA, fc2w, fc2b, out, N_NODES);
}
